// CLSAwareFFN_4260607558028
// MI455X (gfx1250) — matrix operand order repaired and checked by reading, not compiled, not run on hardware
//
#include <hip/hip_runtime.h>

typedef __bf16 bf16_t;
typedef __attribute__((ext_vector_type(16))) __bf16 v16bf;
typedef __attribute__((ext_vector_type(8)))  float  v8f;
typedef __attribute__((ext_vector_type(4)))  int    v4i;

// ---------------------------------------------------------------------------
// CDNA5 async global->LDS copy path (ASYNCcnt-tracked), with sync fallback
// ---------------------------------------------------------------------------
#if defined(__has_builtin)
#if __has_builtin(__builtin_amdgcn_global_load_async_to_lds_b128)
#define USE_ASYNC_LDS 1
#endif
#endif
#ifndef USE_ASYNC_LDS
#define USE_ASYNC_LDS 0
#endif

#if USE_ASYNC_LDS
// builtin signature (from hipcc diagnostic): (v4i AS1*, v4i AS3*, imm i32, imm i32)
typedef __attribute__((address_space(1))) v4i* as1_v4i_ptr;
typedef __attribute__((address_space(3))) v4i* as3_v4i_ptr;

__device__ __forceinline__ void async_copy16(const bf16_t* gsrc, bf16_t* ldst)
{
    __builtin_amdgcn_global_load_async_to_lds_b128(
        (as1_v4i_ptr)gsrc, (as3_v4i_ptr)ldst, /*offset=*/0, /*cpol=*/0);
}
__device__ __forceinline__ void wait_async()
{
#if __has_builtin(__builtin_amdgcn_s_wait_asynccnt)
    __builtin_amdgcn_s_wait_asynccnt(0);
#else
    asm volatile("s_wait_asynccnt 0" ::: "memory");
#endif
}
#endif

// ---------------------------------------------------------------------------
// fp32 -> bf16 conversion, 8 elements / thread (all tensor sizes divide 2048)
// ---------------------------------------------------------------------------
__global__ __launch_bounds__(256)
void cvt_f32_bf16(const float* __restrict__ src, bf16_t* __restrict__ dst, size_t n)
{
    size_t i = ((size_t)blockIdx.x * 256 + threadIdx.x) * 8;
    if (i + 8 <= n) {
        float4 a = *(const float4*)(src + i);
        float4 b = *(const float4*)(src + i + 4);
        union { bf16_t h[8]; uint4 q; } u;
        u.h[0] = (bf16_t)a.x; u.h[1] = (bf16_t)a.y;
        u.h[2] = (bf16_t)a.z; u.h[3] = (bf16_t)a.w;
        u.h[4] = (bf16_t)b.x; u.h[5] = (bf16_t)b.y;
        u.h[6] = (bf16_t)b.z; u.h[7] = (bf16_t)b.w;
        *(uint4*)(dst + i) = u.q;
    }
}

// ---------------------------------------------------------------------------
// WMMA helpers (CDNA5 wave32, 16x16x32 bf16, fp32 accumulate)
// ---------------------------------------------------------------------------
__device__ __forceinline__ v8f wmma_bf16(v16bf a, v16bf b, v8f c)
{
    return __builtin_amdgcn_wmma_f32_16x16x32_bf16(
        false, a, false, b, (short)0, c, false, false);
}

// A fragment 16x32 bf16 (ISA 7.12.2): lanes 0-15 -> row=lane, K 0..7 & 16..23;
// lanes 16-31 -> row=lane-16, K 8..15 & 24..31
__device__ __forceinline__ v16bf load_frag_a(const bf16_t* lds, int lda, int lane)
{
    const int r  = lane & 15;
    const int kq = (lane >> 4) << 3;
    const bf16_t* p = lds + r * lda + kq;
    union { v16bf v; uint4 q[2]; } u;
    u.q[0] = *(const uint4*)(p);
    u.q[1] = *(const uint4*)(p + 16);
    return u.v;
}

// B fragment 32x16 bf16 (W row-major [N,K]): lanes 0-15 -> n=lane, K 0..7 & 16..23;
// lanes 16-31 -> n=lane-16, K 8..15 & 24..31
__device__ __forceinline__ v16bf load_frag_b(const bf16_t* lds, int ldb, int lane)
{
    const int n  = lane & 15;
    const int ko = (lane >> 4) << 3;
    const bf16_t* p = lds + n * ldb + ko;
    union { v16bf v; uint4 q[2]; } u;
    u.q[0] = *(const uint4*)(p);
    u.q[1] = *(const uint4*)(p + 16);
    return u.v;
}

// ---------------------------------------------------------------------------
// Plain GEMM:  C[M,N] = A[M,K] * B[N,K]^T   (optional fused ReLU)
// ---------------------------------------------------------------------------
template<int BM, int BN, int WAVES_M, int WAVES_N, bool RELU>
__global__ __launch_bounds__(WAVES_M * WAVES_N * 32)
void gemm_bf16_wmma(const bf16_t* __restrict__ A,
                    const bf16_t* __restrict__ Bmat,
                    float* __restrict__ C,
                    int M, int N, int K)
{
    constexpr int THREADS = WAVES_M * WAVES_N * 32;
    constexpr int MT  = BM / (WAVES_M * 16);
    constexpr int NT  = BN / (WAVES_N * 16);
    constexpr int KT  = 32;
    constexpr int LDT = KT + 8;
    constexpr int CA  = BM * 4;            // 16-byte chunks per stage
    constexpr int CB  = BN * 4;
    constexpr int TOTC = CA + CB;
    constexpr int CPT  = TOTC / THREADS;
    static_assert(TOTC % THREADS == 0, "chunk count must divide threads");

    __shared__ __align__(16) bf16_t sA[2][BM * LDT];
    __shared__ __align__(16) bf16_t sB[2][BN * LDT];

    const int tid  = threadIdx.x;
    const int lane = tid & 31;
    const int wid  = tid >> 5;
    const int wm   = wid % WAVES_M;
    const int wn   = wid / WAVES_M;
    const int m0   = blockIdx.x * BM;
    const int n0   = blockIdx.y * BN;

    // -------- per-thread copy descriptors (computed once) --------
    const bf16_t* gp[CPT];
    bf16_t* lp0[CPT];
    bf16_t* lp1[CPT];
#pragma unroll
    for (int c = 0; c < CPT; ++c) {
        const int idx = tid + c * THREADS;
        if (idx < CA) {
            const int row = idx >> 2, cc = idx & 3;
            gp[c]  = A + (size_t)(m0 + row) * K + cc * 8;
            lp0[c] = &sA[0][row * LDT + cc * 8];
            lp1[c] = &sA[1][row * LDT + cc * 8];
        } else {
            const int j = idx - CA;
            const int row = j >> 2, cc = j & 3;
            gp[c]  = Bmat + (size_t)(n0 + row) * K + cc * 8;
            lp0[c] = &sB[0][row * LDT + cc * 8];
            lp1[c] = &sB[1][row * LDT + cc * 8];
        }
    }

    v8f acc[MT][NT];
#pragma unroll
    for (int i = 0; i < MT; ++i)
#pragma unroll
        for (int j = 0; j < NT; ++j) acc[i][j] = {};

    auto compute = [&](int buf) {
        v16bf af[MT], bfr[NT];
#pragma unroll
        for (int i = 0; i < MT; ++i)
            af[i] = load_frag_a(&sA[buf][(wm * MT + i) * 16 * LDT], LDT, lane);
#pragma unroll
        for (int j = 0; j < NT; ++j)
            bfr[j] = load_frag_b(&sB[buf][(wn * NT + j) * 16 * LDT], LDT, lane);
#pragma unroll
        for (int i = 0; i < MT; ++i)
#pragma unroll
            for (int j = 0; j < NT; ++j)
                acc[i][j] = wmma_bf16(af[i], bfr[j], acc[i][j]);
    };

    const int KTILES = K / KT;

#if USE_ASYNC_LDS
    auto issue = [&](int buf) {
#pragma unroll
        for (int c = 0; c < CPT; ++c) {
            async_copy16(gp[c], buf ? lp1[c] : lp0[c]);
            gp[c] += KT;
        }
    };
    issue(0);
    wait_async();
    __syncthreads();
    for (int kt = 0; kt < KTILES - 1; ++kt) {
        issue((kt + 1) & 1);
        compute(kt & 1);
        wait_async();
        __syncthreads();
    }
    compute((KTILES - 1) & 1);
#else
    uint4 regs[CPT];
    auto stage = [&]() {
#pragma unroll
        for (int c = 0; c < CPT; ++c) {
            regs[c] = *(const uint4*)gp[c];
            __builtin_prefetch((const void*)(gp[c] + 2 * KT), 0, 0);
            gp[c] += KT;
        }
    };
    auto commit = [&](int buf) {
#pragma unroll
        for (int c = 0; c < CPT; ++c)
            *(uint4*)(buf ? lp1[c] : lp0[c]) = regs[c];
    };
    stage();
    commit(0);
    __syncthreads();
    for (int kt = 0; kt < KTILES - 1; ++kt) {
        stage();                   // global loads for tile kt+1
        compute(kt & 1);
        commit((kt + 1) & 1);      // other buffer: no conflict with readers
        __syncthreads();
    }
    compute((KTILES - 1) & 1);
#endif

    // C/D layout: VGPR e, lanes 0-15 -> M=e, lanes 16-31 -> M=8+e; col = lane&15
    const int row_sel = (lane >> 4) << 3;
    const int col_sel = lane & 15;
#pragma unroll
    for (int i = 0; i < MT; ++i) {
        const int mBase = m0 + (wm * MT + i) * 16 + row_sel;
#pragma unroll
        for (int j = 0; j < NT; ++j) {
            const int nIdx = n0 + (wn * NT + j) * 16 + col_sel;
#pragma unroll
            for (int e = 0; e < 8; ++e) {
                float v = acc[i][j][e];
                if (RELU) v = fmaxf(v, 0.0f);
                C[(size_t)(mBase + e) * N + nIdx] = v;
            }
        }
    }
}

// ---------------------------------------------------------------------------
// Fused gate/up:  Act[M,N] = bf16( silu(X*Wg^T) * (X*Wu^T) )
// ---------------------------------------------------------------------------
template<int BM, int BN, int WAVES_M, int WAVES_N>
__global__ __launch_bounds__(WAVES_M * WAVES_N * 32)
void gate_up_silu_wmma(const bf16_t* __restrict__ X,
                       const bf16_t* __restrict__ Wg,
                       const bf16_t* __restrict__ Wu,
                       bf16_t* __restrict__ Act,
                       int M, int N, int K)
{
    constexpr int THREADS = WAVES_M * WAVES_N * 32;
    constexpr int MT  = BM / (WAVES_M * 16);
    constexpr int NT  = BN / (WAVES_N * 16);
    constexpr int KT  = 32;
    constexpr int LDT = KT + 8;
    constexpr int CA  = BM * 4;
    constexpr int CB  = BN * 4;
    constexpr int TOTC = CA + 2 * CB;
    constexpr int CPT  = TOTC / THREADS;
    static_assert(TOTC % THREADS == 0, "chunk count must divide threads");

    __shared__ __align__(16) bf16_t sA [2][BM * LDT];
    __shared__ __align__(16) bf16_t sBg[2][BN * LDT];
    __shared__ __align__(16) bf16_t sBu[2][BN * LDT];

    const int tid  = threadIdx.x;
    const int lane = tid & 31;
    const int wid  = tid >> 5;
    const int wm   = wid % WAVES_M;
    const int wn   = wid / WAVES_M;
    const int m0   = blockIdx.x * BM;
    const int n0   = blockIdx.y * BN;

    const bf16_t* gp[CPT];
    bf16_t* lp0[CPT];
    bf16_t* lp1[CPT];
#pragma unroll
    for (int c = 0; c < CPT; ++c) {
        const int idx = tid + c * THREADS;
        if (idx < CA) {
            const int row = idx >> 2, cc = idx & 3;
            gp[c]  = X + (size_t)(m0 + row) * K + cc * 8;
            lp0[c] = &sA[0][row * LDT + cc * 8];
            lp1[c] = &sA[1][row * LDT + cc * 8];
        } else if (idx < CA + CB) {
            const int j = idx - CA;
            const int row = j >> 2, cc = j & 3;
            gp[c]  = Wg + (size_t)(n0 + row) * K + cc * 8;
            lp0[c] = &sBg[0][row * LDT + cc * 8];
            lp1[c] = &sBg[1][row * LDT + cc * 8];
        } else {
            const int j = idx - CA - CB;
            const int row = j >> 2, cc = j & 3;
            gp[c]  = Wu + (size_t)(n0 + row) * K + cc * 8;
            lp0[c] = &sBu[0][row * LDT + cc * 8];
            lp1[c] = &sBu[1][row * LDT + cc * 8];
        }
    }

    v8f accG[MT][NT], accU[MT][NT];
#pragma unroll
    for (int i = 0; i < MT; ++i)
#pragma unroll
        for (int j = 0; j < NT; ++j) { accG[i][j] = {}; accU[i][j] = {}; }

    auto compute = [&](int buf) {
        v16bf af[MT], bg[NT], bu[NT];
#pragma unroll
        for (int i = 0; i < MT; ++i)
            af[i] = load_frag_a(&sA[buf][(wm * MT + i) * 16 * LDT], LDT, lane);
#pragma unroll
        for (int j = 0; j < NT; ++j) {
            bg[j] = load_frag_b(&sBg[buf][(wn * NT + j) * 16 * LDT], LDT, lane);
            bu[j] = load_frag_b(&sBu[buf][(wn * NT + j) * 16 * LDT], LDT, lane);
        }
#pragma unroll
        for (int i = 0; i < MT; ++i)
#pragma unroll
            for (int j = 0; j < NT; ++j) {
                accG[i][j] = wmma_bf16(af[i], bg[j], accG[i][j]);
                accU[i][j] = wmma_bf16(af[i], bu[j], accU[i][j]);
            }
    };

    const int KTILES = K / KT;

#if USE_ASYNC_LDS
    auto issue = [&](int buf) {
#pragma unroll
        for (int c = 0; c < CPT; ++c) {
            async_copy16(gp[c], buf ? lp1[c] : lp0[c]);
            gp[c] += KT;
        }
    };
    issue(0);
    wait_async();
    __syncthreads();
    for (int kt = 0; kt < KTILES - 1; ++kt) {
        issue((kt + 1) & 1);
        compute(kt & 1);
        wait_async();
        __syncthreads();
    }
    compute((KTILES - 1) & 1);
#else
    uint4 regs[CPT];
    auto stage = [&]() {
#pragma unroll
        for (int c = 0; c < CPT; ++c) {
            regs[c] = *(const uint4*)gp[c];
            __builtin_prefetch((const void*)(gp[c] + 2 * KT), 0, 0);
            gp[c] += KT;
        }
    };
    auto commit = [&](int buf) {
#pragma unroll
        for (int c = 0; c < CPT; ++c)
            *(uint4*)(buf ? lp1[c] : lp0[c]) = regs[c];
    };
    stage();
    commit(0);
    __syncthreads();
    for (int kt = 0; kt < KTILES - 1; ++kt) {
        stage();
        compute(kt & 1);
        commit((kt + 1) & 1);
        __syncthreads();
    }
    compute((KTILES - 1) & 1);
#endif

    const int row_sel = (lane >> 4) << 3;
    const int col_sel = lane & 15;
#pragma unroll
    for (int i = 0; i < MT; ++i) {
        const int mBase = m0 + (wm * MT + i) * 16 + row_sel;
#pragma unroll
        for (int j = 0; j < NT; ++j) {
            const int nIdx = n0 + (wn * NT + j) * 16 + col_sel;
#pragma unroll
            for (int e = 0; e < 8; ++e) {
                const float g  = accG[i][j][e];
                const float u  = accU[i][j][e];
                const float sg = 1.0f / (1.0f + __builtin_amdgcn_exp2f(-g * 1.44269504088896f));
                Act[(size_t)(mBase + e) * N + nIdx] = (bf16_t)(g * sg * u);
            }
        }
    }
}

// ---------------------------------------------------------------------------
// Host-side launcher
// ---------------------------------------------------------------------------
extern "C" void kernel_launch(void* const* d_in, const int* in_sizes, int n_in,
                              void* d_out, int out_size, void* d_ws, size_t ws_size,
                              hipStream_t stream)
{
    (void)in_sizes; (void)n_in; (void)out_size; (void)ws_size;

    constexpr int Bz = 4, S = 2048, H = 2048, I = 8192, E = 64;
    constexpr int M  = Bz * S;                    // 8192 token rows

    const float* X  = (const float*)d_in[0];      // [B,S,H]
    const float* Wr = (const float*)d_in[1];      // [E,H]
    const float* Wg = (const float*)d_in[2];      // [I,H]
    const float* Wu = (const float*)d_in[3];      // [I,H]
    const float* Wd = (const float*)d_in[4];      // [H,I]

    float* out_ffn    = (float*)d_out;                      // [B,S,H]
    float* out_router = (float*)d_out + (size_t)M * H;      // [B,S,E]

    bf16_t* ws  = (bf16_t*)d_ws;
    size_t off  = 0;
    bf16_t* Xb  = ws + off; off += (size_t)M * H;
    bf16_t* Wgb = ws + off; off += (size_t)I * H;
    bf16_t* Wub = ws + off; off += (size_t)I * H;
    bf16_t* Wdb = ws + off; off += (size_t)H * I;
    bf16_t* Wrb = ws + off; off += (size_t)E * H;
    bf16_t* Act = ws + off;                                 // [M, I]

    auto cvt = [&](const float* s, bf16_t* d, size_t n) {
        cvt_f32_bf16<<<dim3((unsigned)(n / 2048)), dim3(256), 0, stream>>>(s, d, n);
    };
    cvt(X,  Xb,  (size_t)M * H);
    cvt(Wg, Wgb, (size_t)I * H);
    cvt(Wu, Wub, (size_t)I * H);
    cvt(Wd, Wdb, (size_t)H * I);
    cvt(Wr, Wrb, (size_t)E * H);

    // router: [M,H] x [E,H]^T -> ReLU
    gemm_bf16_wmma<128, 64, 4, 2, true>
        <<<dim3(M / 128, E / 64), dim3(256), 0, stream>>>(Xb, Wrb, out_router, M, E, H);

    // fused gate/up + SiLU -> Act (bf16)
    gate_up_silu_wmma<128, 64, 4, 2>
        <<<dim3(M / 128, I / 64), dim3(256), 0, stream>>>(Xb, Wgb, Wub, Act, M, I, H);

    // down: [M,I] x [H,I]^T -> out
    gemm_bf16_wmma<128, 128, 4, 2, false>
        <<<dim3(M / 128, H / 128), dim3(256), 0, stream>>>(Act, Wdb, out_ffn, M, H, I);
}
